// ELMO_55284819034588
// MI455X (gfx1250) — compile-verified, matrix operand order checked by reading, not run on hardware
//
#include <hip/hip_runtime.h>
#include <stdint.h>

// ELMo-style pipeline for MI455X (gfx1250, wave32, WMMA + async LDS DMA).
// B=32 T=128 V=32000 E=256 H=256 L=2 ; S=T-1=127 steps, ctx rows M=32*126=4032.

#define DEVI __device__ __forceinline__

typedef __attribute__((ext_vector_type(16))) __bf16 v16bf;
typedef __attribute__((ext_vector_type(8)))  float  v8f;

union FragU { uint4 q[2]; v16bf v; };

DEVI unsigned short f2bf(float f) {
  unsigned int u = __float_as_uint(f);
  u += 0x7FFFu + ((u >> 16) & 1u);
  return (unsigned short)(u >> 16);
}
DEVI float bf2f(unsigned short s) { return __uint_as_float(((unsigned int)s) << 16); }
DEVI float sigm(float x) { return 1.0f / (1.0f + __expf(-x)); }

DEVI v8f wmma_bf16_acc(const FragU& A, const FragU& B, v8f c) {
  // v_wmma_f32_16x16x32_bf16  (neg_a, A, neg_b, B, c_mod, C, reuse_a, reuse_b)
  return __builtin_amdgcn_wmma_f32_16x16x32_bf16(false, A.v, false, B.v, (short)0, c, false, false);
}

// CDNA5 async DMA: copy 32 bytes/lane global->LDS without touching VGPR data.
// INST_OFFSET is added to both LDS and global addresses (ISA 08 sec 4.4), so one
// LDS-address VGPR covers both halves. Tracked by ASYNCcnt.
DEVI void async_stage_b32B(const unsigned short* g, unsigned short* l) {
  unsigned lds = (unsigned)(uintptr_t)l;   // generic->LDS: low 32 bits are the LDS address
  asm volatile("global_load_async_to_lds_b128 %0, %1, off"
               :: "v"(lds), "v"(g) : "memory");
  asm volatile("global_load_async_to_lds_b128 %0, %1, off offset:16"
               :: "v"(lds), "v"(g) : "memory");
}
DEVI void wait_async_all() { asm volatile("s_wait_asynccnt 0x0" ::: "memory"); }

// B-operand swizzle: per 32x16 (KxN) tile store 16 contiguous bf16 per lane so a
// B fragment is one 32-byte load. Lane layout per ISA 16-bit operand tables:
// lane = (n&15) + 16*half, half selects K groups {0-7,16-23} vs {8-15,24-31}.
DEVI size_t bswz_idx(int k, int n, int nblocks) {
  int kin  = k & 31;
  int half = (kin & 15) >> 3;
  int e    = (kin & 7) + ((kin >> 4) << 3);
  int lane = (n & 15) + (half << 4);
  return ((size_t)((k >> 5) * nblocks + (n >> 4)) * 32 + lane) * 16 + e;
}

// ---------------------------------------------------------------- embed gather
__global__ void embed_kernel(const int* __restrict__ x, const float* __restrict__ table,
                             unsigned short* __restrict__ emb) {
  int bt = blockIdx.x;                 // 0..4095  (b*128+t)
  int e  = threadIdx.x;                // 0..255
  int tok = x[bt];
  emb[(size_t)bt * 256 + e] = f2bf(table[(size_t)tok * 256 + e]);
}

// ------------------------------------------------- weight f32 -> bf16 swizzle
__global__ void swizzle_kernel(const float* __restrict__ src, unsigned short* __restrict__ dst,
                               int K, int N) {
  size_t idx = (size_t)blockIdx.x * blockDim.x + threadIdx.x;
  size_t total = (size_t)K * N;
  if (idx >= total) return;
  int k = (int)(idx / N), n = (int)(idx % N);
  dst[bswz_idx(k, n, N >> 4)] = f2bf(src[idx]);
}

// --------------------------------- XW0 = emb(dir-mapped) @ W0_bf16 + b0  (WMMA)
// rows m = b*127 + step (4064 rows = 254 tiles); forward srcT=step, backward srcT=127-step
__global__ __launch_bounds__(256) void xw0_kernel(
    const unsigned short* __restrict__ emb, const unsigned short* __restrict__ wsw,
    const float* __restrict__ fb, const float* __restrict__ bb, float* __restrict__ xw0) {
  const int dir  = blockIdx.z;
  const int mt   = blockIdx.y;                  // 0..253
  const int wave = threadIdx.x >> 5, lane = threadIdx.x & 31;
  const int nb   = blockIdx.x * 8 + wave;       // 0..63
  const unsigned short* W0 = wsw + ((size_t)(dir ? 4 : 0)) * 262144;
  const float* bias = dir ? bb : fb;            // layer 0 bias
  const int n  = nb * 16 + (lane & 15);
  const float bv = bias[n];
  v8f c;
#pragma unroll
  for (int r = 0; r < 8; ++r) c[r] = bv;

  const int rowA = mt * 16 + (lane & 15);
  const int b = rowA / 127, step = rowA % 127;
  const int srcT = dir ? (127 - step) : step;
  const unsigned short* arow = emb + ((size_t)(b * 128 + srcT)) * 256;
  const int ah = (lane >> 4) << 3;
#pragma unroll
  for (int ks = 0; ks < 8; ++ks) {
    FragU A, Bf;
    A.q[0] = reinterpret_cast<const uint4*>(arow + ks * 32 + ah)[0];
    A.q[1] = reinterpret_cast<const uint4*>(arow + ks * 32 + 16 + ah)[0];
    const uint4* bp2 = reinterpret_cast<const uint4*>(W0 + ((size_t)(ks * 64 + nb) * 32 + lane) * 16);
    Bf.q[0] = bp2[0]; Bf.q[1] = bp2[1];
    c = wmma_bf16_acc(A, Bf, c);
  }
  float* dst = xw0 + (size_t)dir * 4064 * 1024;
  const int m0 = mt * 16 + ((lane >> 4) << 3);
#pragma unroll
  for (int r = 0; r < 8; ++r) dst[(size_t)(m0 + r) * 1024 + n] = c[r];
}

// ----------------------------- recurrent helper: cc += h(LDS 32x256) @ B(swz)
DEVI void gemm8_acc(v8f (&cc)[2][2], const unsigned short* hsrc,
                    const unsigned short* Bmat, int nb0, int lane) {
  const int ah = (lane >> 4) << 3;
  const int arow0 = lane & 15;
#pragma unroll
  for (int ks = 0; ks < 8; ++ks) {
    FragU A[2];
#pragma unroll
    for (int mi = 0; mi < 2; ++mi) {
      const unsigned short* ar = hsrc + (arow0 + mi * 16) * 256 + ks * 32 + ah;
      A[mi].q[0] = reinterpret_cast<const uint4*>(ar)[0];
      A[mi].q[1] = reinterpret_cast<const uint4*>(ar + 16)[0];
    }
#pragma unroll
    for (int j = 0; j < 2; ++j) {
      FragU Bf;
      const uint4* bp2 = reinterpret_cast<const uint4*>(Bmat + ((size_t)(ks * 64 + nb0 + j) * 32 + lane) * 16);
      Bf.q[0] = bp2[0]; Bf.q[1] = bp2[1];
      cc[0][j] = wmma_bf16_acc(A[0], Bf, cc[0][j]);
      cc[1][j] = wmma_bf16_acc(A[1], Bf, cc[1][j]);
    }
  }
}

// ------------------------------- persistent LSTM: 1 block per direction, 32 waves
__global__ __launch_bounds__(1024) void lstm_kernel(
    const int* __restrict__ x, const float* __restrict__ xw0,
    const unsigned short* __restrict__ wsw,
    const float* __restrict__ fb, const float* __restrict__ bb,
    float* __restrict__ ctx_f32, unsigned short* __restrict__ ctx_bf16) {
  __shared__ __align__(16) float zbuf[32 * 1024];                 // 128 KB
  __shared__ __align__(16) unsigned short hbuf[2][2][32 * 256];   // 64 KB (layer, pingpong)

  const int dir  = blockIdx.x;
  const int tid  = threadIdx.x;
  const int wave = tid >> 5, lane = tid & 31;
  const int nb0  = wave * 2;

  const unsigned short* U0 = wsw + ((size_t)(dir ? 6 : 2)) * 262144;
  const unsigned short* W1 = wsw + ((size_t)(dir ? 5 : 1)) * 262144;
  const unsigned short* U1 = wsw + ((size_t)(dir ? 7 : 3)) * 262144;
  const float* bias1 = (dir ? bb : fb) + 1024;
  const float* xw = xw0 + (size_t)dir * 4064 * 1024;

  float c0r[8], c1r[8];
#pragma unroll
  for (int i = 0; i < 8; ++i) c0r[i] = c1r[i] = 0.0f;
  for (int i = tid; i < 32 * 256; i += 1024) { hbuf[0][0][i] = 0; hbuf[1][0][i] = 0; }
  __syncthreads();

  const int nlo = lane & 15;
  const int n0 = nb0 * 16 + nlo, n1 = n0 + 16;
  const int mBase = (lane >> 4) << 3;

  for (int t = 0; t < 127; ++t) {
    const int prev = t & 1, cur = prev ^ 1;
    const int xt = dir ? (127 - t) : t;

    { // ---- layer 0: z = XW0[:,t,:] + h0_prev @ U0 ----
      v8f cc[2][2];
#pragma unroll
      for (int mi = 0; mi < 2; ++mi)
#pragma unroll
        for (int r = 0; r < 8; ++r) {
          int m = mi * 16 + mBase + r;
          size_t ro = ((size_t)m * 127 + t) * 1024;
          cc[mi][0][r] = xw[ro + n0];
          cc[mi][1][r] = xw[ro + n1];
        }
      gemm8_acc(cc, &hbuf[0][prev][0], U0, nb0, lane);
#pragma unroll
      for (int mi = 0; mi < 2; ++mi)
#pragma unroll
        for (int r = 0; r < 8; ++r) {
          int m = mi * 16 + mBase + r;
          zbuf[m * 1024 + n0] = cc[mi][0][r];
          zbuf[m * 1024 + n1] = cc[mi][1][r];
        }
    }
    __syncthreads();

    // ---- layer 0 gates (i,f,g,o), mask carry ----
#pragma unroll
    for (int i = 0; i < 8; ++i) {
      int id = tid + (i << 10);
      int m = id >> 8, hid = id & 255;
      float zi = zbuf[m * 1024 + hid];
      float zf = zbuf[m * 1024 + 256 + hid];
      float zg = zbuf[m * 1024 + 512 + hid];
      float zo = zbuf[m * 1024 + 768 + hid];
      float cp = c0r[i];
      float cn = sigm(zf) * cp + sigm(zi) * tanhf(zg);
      float hn = sigm(zo) * tanhf(cn);
      if (x[m * 128 + xt] == 0) { hn = bf2f(hbuf[0][prev][m * 256 + hid]); cn = cp; }
      c0r[i] = cn;
      hbuf[0][cur][m * 256 + hid] = f2bf(hn);
    }
    __syncthreads();

    { // ---- layer 1: z = h0_cur @ W1 + h1_prev @ U1 + b1 ----
      v8f cc[2][2];
      float bv0 = bias1[n0], bv1 = bias1[n1];
#pragma unroll
      for (int r = 0; r < 8; ++r) { cc[0][0][r] = bv0; cc[1][0][r] = bv0; cc[0][1][r] = bv1; cc[1][1][r] = bv1; }
      gemm8_acc(cc, &hbuf[0][cur][0], W1, nb0, lane);
      gemm8_acc(cc, &hbuf[1][prev][0], U1, nb0, lane);
#pragma unroll
      for (int mi = 0; mi < 2; ++mi)
#pragma unroll
        for (int r = 0; r < 8; ++r) {
          int m = mi * 16 + mBase + r;
          zbuf[m * 1024 + n0] = cc[mi][0][r];
          zbuf[m * 1024 + n1] = cc[mi][1][r];
        }
    }
    __syncthreads();

    // ---- layer 1 gates + ctx writeback ----
    const bool doStore = dir ? (t >= 1) : (t < 126);
    const int pos  = dir ? (t - 1) : t;
    const int half = dir ? 256 : 0;
#pragma unroll
    for (int i = 0; i < 8; ++i) {
      int id = tid + (i << 10);
      int m = id >> 8, hid = id & 255;
      float zi = zbuf[m * 1024 + hid];
      float zf = zbuf[m * 1024 + 256 + hid];
      float zg = zbuf[m * 1024 + 512 + hid];
      float zo = zbuf[m * 1024 + 768 + hid];
      float cp = c1r[i];
      float cn = sigm(zf) * cp + sigm(zi) * tanhf(zg);
      float hn = sigm(zo) * tanhf(cn);
      if (x[m * 128 + xt] == 0) { hn = bf2f(hbuf[1][prev][m * 256 + hid]); cn = cp; }
      c1r[i] = cn;
      hbuf[1][cur][m * 256 + hid] = f2bf(hn);
      if (doStore) {
        size_t ci = ((size_t)m * 126 + pos) * 512 + half + hid;
        ctx_f32[ci] = hn;
        ctx_bf16[ci] = f2bf(hn);
      }
    }
    __syncthreads();
  }
}

// -------------------- projection: out = ctx[4032,512] @ Wp[512,32000] + bp ----
// 128Mx128N block tile, 8 waves (wave = 16Mx128N strip).
// B chunks (8KB) are DMA'd into LDS with global_load_async_to_lds_b128
// (ASYNCcnt), double-buffered and overlapped with the WMMA K-loop.
__global__ __launch_bounds__(256) void proj_kernel(
    const unsigned short* __restrict__ ctxb, const unsigned short* __restrict__ wpsw,
    const float* __restrict__ bp, float* __restrict__ out) {
  __shared__ __align__(16) unsigned short bstage[2][4096];   // 2 x 8KB
  const int tid = threadIdx.x, wave = tid >> 5, lane = tid & 31;
  const int nbBase = blockIdx.x * 8;                 // 16-col tile units (0..1992)
  const int mStrip = blockIdx.y * 128 + wave * 16;
  const bool mOK = (mStrip < 4032);
  const int arow = mStrip + (lane & 15);
  const int ah = (lane >> 4) << 3;

  // stage k-step 0 via async DMA
  async_stage_b32B(wpsw + (size_t)nbBase * 512 + (size_t)tid * 16, &bstage[0][tid * 16]);
  wait_async_all();
  __syncthreads();

  v8f cc[8];
#pragma unroll
  for (int j = 0; j < 8; ++j)
#pragma unroll
    for (int r = 0; r < 8; ++r) cc[j][r] = 0.0f;

  for (int ks = 0; ks < 16; ++ks) {
    const int buf = ks & 1;
    const bool more = (ks + 1 < 16);
    if (more)   // kick off DMA of the next 8KB B chunk into the other buffer
      async_stage_b32B(wpsw + ((size_t)(ks + 1) * 2000 + nbBase) * 512 + (size_t)tid * 16,
                       &bstage[buf ^ 1][tid * 16]);
    if (ks + 2 < 16)   // global_prefetch_b8 of the chunk after next (L2 warm)
      __builtin_prefetch(wpsw + ((size_t)(ks + 2) * 2000 + nbBase) * 512 + (size_t)tid * 16, 0, 0);

    if (mOK) {
      FragU A;
      const unsigned short* ar = ctxb + (size_t)arow * 512 + ks * 32 + ah;
      A.q[0] = reinterpret_cast<const uint4*>(ar)[0];
      A.q[1] = reinterpret_cast<const uint4*>(ar + 16)[0];
#pragma unroll
      for (int j = 0; j < 8; ++j) {
        FragU Bf;
        const uint4* bp2 = reinterpret_cast<const uint4*>(&bstage[buf][(j * 32 + lane) * 16]);
        Bf.q[0] = bp2[0]; Bf.q[1] = bp2[1];
        cc[j] = wmma_bf16_acc(A, Bf, cc[j]);
      }
    }
    if (more) wait_async_all();   // this wave's DMA done; barrier makes it WG-wide
    __syncthreads();
  }

  if (mOK) {
    const int mrow0 = mStrip + ((lane >> 4) << 3);
#pragma unroll
    for (int j = 0; j < 8; ++j) {
      int n = (nbBase + j) * 16 + (lane & 15);
      float bias = bp[n];
#pragma unroll
      for (int r = 0; r < 8; ++r)
        out[(size_t)(mrow0 + r) * 32000 + n] = cc[j][r] + bias;
    }
  }
}

// ------------------------------------------ NSP head: ctx.reshape(32,-1) @ Wn
__global__ void nsp_kernel(const float* __restrict__ ctx, const float* __restrict__ Wn,
                           const float* __restrict__ bn, float* __restrict__ out) {
  int o = blockIdx.x;             // 0..63 = b*2 + j
  int b = o >> 1, j = o & 1;
  const float* row = ctx + (size_t)b * 64512;
  float s = 0.0f;
  for (int i = threadIdx.x; i < 64512; i += 256) s += row[i] * Wn[(size_t)i * 2 + j];
  __shared__ float red[256];
  red[threadIdx.x] = s;
  __syncthreads();
  for (int off = 128; off; off >>= 1) {
    if (threadIdx.x < off) red[threadIdx.x] += red[threadIdx.x + off];
    __syncthreads();
  }
  if (threadIdx.x == 0) out[129024000u + o] = red[0] + bn[j];
}

// ---------------------------------------------------------------------- launch
extern "C" void kernel_launch(void* const* d_in, const int* in_sizes, int n_in,
                              void* d_out, int out_size, void* d_ws, size_t ws_size,
                              hipStream_t stream) {
  (void)in_sizes; (void)n_in; (void)out_size; (void)ws_size;
  const int*   x     = (const int*)d_in[0];
  const float* table = (const float*)d_in[1];
  const float* fW = (const float*)d_in[2];
  const float* fU = (const float*)d_in[3];
  const float* fb = (const float*)d_in[4];
  const float* bW = (const float*)d_in[5];
  const float* bU = (const float*)d_in[6];
  const float* bb = (const float*)d_in[7];
  const float* Wp = (const float*)d_in[8];
  const float* bp = (const float*)d_in[9];
  const float* Wn = (const float*)d_in[10];
  const float* bn = (const float*)d_in[11];
  float* out = (float*)d_out;

  // workspace layout (bytes, 256-aligned); total ~84.8 MB
  char* ws = (char*)d_ws;
  unsigned short* emb_bf = (unsigned short*)(ws + 0);            //  2,097,152
  unsigned short* wsw    = (unsigned short*)(ws + 2097152);      //  4,194,304 (8 x 256x1024 bf16 swz)
  unsigned short* wp_sw  = (unsigned short*)(ws + 6291456);      // 32,768,000
  float*          xw0    = (float*)         (ws + 39059456);     // 33,292,288
  float*          ctxf   = (float*)         (ws + 72351744);     //  8,257,536
  unsigned short* ctxb   = (unsigned short*)(ws + 80609280);     //  4,128,768

  embed_kernel<<<4096, 256, 0, stream>>>(x, table, emb_bf);

  const float* wsrc[8] = { fW, fW + 262144, fU, fU + 262144,
                           bW, bW + 262144, bU, bU + 262144 };
  for (int i = 0; i < 8; ++i)
    swizzle_kernel<<<1024, 256, 0, stream>>>(wsrc[i], wsw + (size_t)i * 262144, 256, 1024);
  swizzle_kernel<<<64000, 256, 0, stream>>>(Wp, wp_sw, 512, 32000);

  xw0_kernel<<<dim3(8, 254, 2), 256, 0, stream>>>(emb_bf, wsw, fb, bb, xw0);
  lstm_kernel<<<2, 1024, 0, stream>>>(x, xw0, wsw, fb, bb, ctxf, ctxb);
  proj_kernel<<<dim3(250, 32), 256, 0, stream>>>(ctxb, wp_sw, bp, out);
  nsp_kernel<<<64, 256, 0, stream>>>(ctxf, Wn, bn, out);
}
